// Trans_high_30279519437480
// MI455X (gfx1250) — matrix operand order repaired and checked by reading, not compiled, not run on hardware
//
#include <hip/hip_runtime.h>

typedef __attribute__((ext_vector_type(16))) _Float16 v16h;
typedef __attribute__((ext_vector_type(8)))  _Float16 v8h;
typedef __attribute__((ext_vector_type(8)))  float    v8f;
typedef __attribute__((ext_vector_type(4)))  unsigned int v4u;
typedef __attribute__((ext_vector_type(8)))  int      v8i;
typedef __attribute__((ext_vector_type(4)))  int      v4i;

#define WMMA16(a, b, c) \
  __builtin_amdgcn_wmma_f32_16x16x32_f16(false, (a), false, (b), (short)0, (c), false, false)

#if defined(__has_builtin)
# if __has_builtin(__builtin_amdgcn_tensor_load_to_lds) && __has_builtin(__builtin_amdgcn_s_wait_tensorcnt)
#  define HAS_TDM 1
# else
#  define HAS_TDM 0
# endif
#else
# define HAS_TDM 0
#endif

// ---------------------------------------------------------------------------
// TDM: DMA a 2D f32 tile (tile_rows x tile_cols, row stride row_stride elems)
// from global memory into LDS at byte offset lds_off. D# built per ISA 8.3/8.4.
// ---------------------------------------------------------------------------
__device__ __forceinline__ void tdm_load_2d_f32(unsigned lds_off, const void* gp,
                                                unsigned tile_cols, unsigned tile_rows,
                                                unsigned long long row_stride,
                                                unsigned td0, unsigned td1)
{
#if HAS_TDM
  unsigned long long ga = (unsigned long long)(size_t)gp;
  v4u g0 = { 1u,                                    // count=1 (valid user D#)
             lds_off,                               // lds_addr (bytes)
             (unsigned)(ga & 0xffffffffu),
             (unsigned)((ga >> 32) & 0x01ffffffu) | (2u << 30) };  // addr[56:32] | type=2
  v8i g1;
  g1[0] = (int)(2u << 16);                          // data_size = 4 bytes
  g1[1] = (int)((td0 & 0xffffu) << 16);             // tensor_dim0[15:0] @ bits 63:48
  g1[2] = (int)((td0 >> 16) | ((td1 & 0xffffu) << 16));
  g1[3] = (int)((td1 >> 16) | ((tile_cols & 0xffffu) << 16));  // tile_dim0
  g1[4] = (int)(tile_rows & 0xffffu);               // tile_dim1 (tile_dim2 = 0)
  g1[5] = (int)(row_stride & 0xffffffffull);        // tensor_dim0_stride lo32
  g1[6] = (int)((row_stride >> 32) & 0xffffull);    // stride hi16; dim1_stride lo16 = 0
  g1[7] = 0;
  v4i gz = {0, 0, 0, 0};
#if defined(__clang_major__) && __clang_major__ >= 23
  v8i gz8 = {0, 0, 0, 0, 0, 0, 0, 0};
  __builtin_amdgcn_tensor_load_to_lds(g0, g1, gz, gz, gz8, 0);
#else
  __builtin_amdgcn_tensor_load_to_lds(g0, g1, gz, gz, 0);
#endif
#else
  (void)lds_off; (void)gp; (void)tile_cols; (void)tile_rows;
  (void)row_stride; (void)td0; (void)td1;
#endif
}

// ---------------------------------------------------------------------------
// Weight repack: w (Cout,Cin,KH,KW) f32 -> wpk[m][tap][ci] f16 with m padded to
// CoutP (mult 32) and ci padded to CinP (mult 32), zero filled.
// ---------------------------------------------------------------------------
__global__ void k_wpack(const float* __restrict__ w, _Float16* __restrict__ o,
                        int Cout, int CoutP, int Cin, int CinP, int KK)
{
  long long i = (long long)blockIdx.x * blockDim.x + threadIdx.x;
  long long tot = (long long)CoutP * KK * CinP;
  if (i >= tot) return;
  int cp = (int)(i % CinP); long long t = i / CinP;
  int tap = (int)(t % KK);
  int m = (int)(t / KK);
  float v = (m < Cout && cp < Cin) ? w[((size_t)m * Cin + cp) * KK + tap] : 0.f;
  o[i] = (_Float16)v;
}

// ---------------------------------------------------------------------------
// Implicit-GEMM convolution (stride 1, SAME), tap-major K ordering, packed f16
// weights.  One wave: 32(Cout) x 64(pixel) tile -> 8 WMMAs per 32-wide K-step.
// ---------------------------------------------------------------------------
__global__ __launch_bounds__(32)
void k_conv_wmma(const float* __restrict__ x, const _Float16* __restrict__ wpk,
                 const float* __restrict__ bias, const float* __restrict__ res,
                 float* __restrict__ y,
                 int B, int Cin, int CinP, int H, int W, int Cout,
                 int KH, int KW, int pad, int act)
{
  const int lane = (int)threadIdx.x;
  const int nl   = lane & 15;
  const int sub  = lane >> 4;
  const int m0   = blockIdx.y * 32;
  const int p0   = blockIdx.x * 64;
  const int HW = H * W;
  const int NP = B * HW;
  const int KK = KH * KW;

  int pnb[4], prem[4], poh[4], pow_[4];
  bool pv[4];
#pragma unroll
  for (int t = 0; t < 4; ++t) {
    int p = p0 + t * 16 + nl;
    pv[t] = (p < NP);
    int pc = pv[t] ? p : 0;
    pnb[t] = pc / HW;
    prem[t] = pc - pnb[t] * HW;
    poh[t] = prem[t] / W;
    pow_[t] = prem[t] - poh[t] * W;
  }
  const _Float16* wr0 = wpk + (size_t)(m0 + nl) * KK * CinP;
  const _Float16* wr1 = wpk + (size_t)(m0 + 16 + nl) * KK * CinP;

  v8f acc[2][4] = {};

  for (int tap = 0; tap < KK; ++tap) {
    const int kh = tap / KW, kw = tap - kh * KW;
    const int dh = kh - pad, dw = kw - pad;
    const float* bb[4];
    bool bval[4];
#pragma unroll
    for (int t = 0; t < 4; ++t) {
      int ih = poh[t] + dh, iw = pow_[t] + dw;
      bval[t] = pv[t] && ih >= 0 && ih < H && iw >= 0 && iw < W;
      long long off = bval[t] ? ((long long)pnb[t] * Cin * HW + (long long)ih * W + iw) : 0ll;
      bb[t] = x + off;
    }
    const _Float16* a0p = wr0 + (size_t)tap * CinP;
    const _Float16* a1p = wr1 + (size_t)tap * CinP;

    for (int c0 = 0; c0 < CinP; c0 += 32) {
      if (c0 + 32 < CinP) __builtin_prefetch(a0p + c0 + 32, 0, 0);
      // A fragments: aligned contiguous v8h loads (padding supplies zeros)
      v8h lo0 = *(const v8h*)(a0p + c0 + sub * 8);
      v8h hi0 = *(const v8h*)(a0p + c0 + 16 + sub * 8);
      v8h lo1 = *(const v8h*)(a1p + c0 + sub * 8);
      v8h hi1 = *(const v8h*)(a1p + c0 + 16 + sub * 8);
      v16h A0 = __builtin_shufflevector(lo0, hi0, 0,1,2,3,4,5,6,7,8,9,10,11,12,13,14,15);
      v16h A1 = __builtin_shufflevector(lo1, hi1, 0,1,2,3,4,5,6,7,8,9,10,11,12,13,14,15);
      // B fragments (stride HW; mask ci >= Cin to avoid NaN-poisoning)
      v16h bf[4];
#pragma unroll
      for (int j = 0; j < 16; ++j) {
        int c = c0 + ((j & 8) << 1) + sub * 8 + (j & 7);
        bool cv = (c < Cin);
        size_t co = (size_t)c * HW;
#pragma unroll
        for (int t = 0; t < 4; ++t)
          bf[t][j] = (_Float16)((cv && bval[t]) ? bb[t][co] : 0.f);
      }
#pragma unroll
      for (int t = 0; t < 4; ++t) {
        acc[0][t] = WMMA16(A0, bf[t], acc[0][t]);
        acc[1][t] = WMMA16(A1, bf[t], acc[1][t]);
      }
    }
  }
  // epilogue
#pragma unroll
  for (int mt = 0; mt < 2; ++mt) {
#pragma unroll
    for (int t = 0; t < 4; ++t) {
      if (!pv[t]) continue;
#pragma unroll
      for (int i = 0; i < 8; ++i) {
        int mm = m0 + mt * 16 + i + 8 * sub;
        if (mm >= Cout) continue;
        float v = acc[mt][t][i];
        if (bias) v += bias[mm];
        if (act == 1) v = (v >= 0.f) ? v : 0.01f * v;
        size_t off = ((size_t)pnb[t] * Cout + mm) * HW + prem[t];
        if (res) v += res[off];
        y[off] = v;
      }
    }
  }
}

// ---------------------------------------------------------------------------
// Batched split-K GEMM (NT): att[z] += q[z](48 x n) x k[z]^T over a K-chunk.
// TDM stages 48x128 tiles of q and k into LDS (s_wait_tensorcnt), fragments
// are then read from LDS; 9 WMMAs per 32-wide K-step.  Atomic accumulation.
// ---------------------------------------------------------------------------
__global__ __launch_bounds__(32)
void k_gemm_nt_wmma(const float* __restrict__ q, const float* __restrict__ kmat,
                    float* __restrict__ att,
                    int heads, int chead, int n,
                    int CqFull, int cq0, int CkFull, int ck0, int nsplit)
{
  __shared__ float ldsQ[48 * 128];
  __shared__ float ldsK[48 * 128];
  const int z = blockIdx.z;
  const int b = z / heads, h = z - b * heads;
  const float* A  = q    + ((size_t)b * CqFull + cq0 + h * chead) * n;
  const float* Bm = kmat + ((size_t)b * CkFull + ck0 + h * chead) * n;
  float* C = att + (size_t)z * chead * chead;
  const int lane = (int)threadIdx.x, nl = lane & 15, sub = lane >> 4;
  const int chunk = n / nsplit;               // multiple of 128 by construction
  const int ks = blockIdx.x * chunk, ke = ks + chunk;

  v8f acc[3][3] = {};
#if HAS_TDM
  const unsigned lq = (unsigned)(size_t)(void*)ldsQ;
  const unsigned lk = (unsigned)(size_t)(void*)ldsK;
  for (int k0 = ks; k0 < ke; k0 += 128) {
    tdm_load_2d_f32(lq, A + k0, 128, 48, (unsigned long long)n, (unsigned)(n - k0), 48u);
    tdm_load_2d_f32(lk, Bm + k0, 128, 48, (unsigned long long)n, (unsigned)(n - k0), 48u);
    __builtin_amdgcn_s_wait_tensorcnt((short)0);
#pragma unroll
    for (int kc = 0; kc < 128; kc += 32) {
      v16h a[3], bf[3];
#pragma unroll
      for (int i = 0; i < 3; ++i) {
        const float* qr = ldsQ + (i * 16 + nl) * 128 + kc;
        const float* kr = ldsK + (i * 16 + nl) * 128 + kc;
#pragma unroll
        for (int j = 0; j < 8; ++j) {
          a[i][j]     = (_Float16)qr[sub * 8 + j];
          a[i][8 + j] = (_Float16)qr[16 + sub * 8 + j];
        }
#pragma unroll
        for (int j = 0; j < 16; ++j)
          bf[i][j] = (_Float16)kr[((j & 8) << 1) + sub * 8 + (j & 7)];
      }
#pragma unroll
      for (int mi = 0; mi < 3; ++mi)
#pragma unroll
        for (int ni = 0; ni < 3; ++ni)
          acc[mi][ni] = WMMA16(a[mi], bf[ni], acc[mi][ni]);
    }
  }
#else
  const float* ar[3];
  const float* br[3];
#pragma unroll
  for (int i = 0; i < 3; ++i) {
    ar[i] = A  + (size_t)(i * 16 + nl) * n;
    br[i] = Bm + (size_t)(i * 16 + nl) * n;
  }
  for (int k0 = ks; k0 < ke; k0 += 32) {
    v16h a[3], bf[3];
#pragma unroll
    for (int i = 0; i < 3; ++i) {
#pragma unroll
      for (int j = 0; j < 8; ++j) {
        a[i][j]     = (_Float16)ar[i][k0 + sub * 8 + j];
        a[i][8 + j] = (_Float16)ar[i][k0 + 16 + sub * 8 + j];
      }
#pragma unroll
      for (int j = 0; j < 16; ++j)
        bf[i][j] = (_Float16)br[i][k0 + ((j & 8) << 1) + sub * 8 + (j & 7)];
    }
#pragma unroll
    for (int mi = 0; mi < 3; ++mi)
#pragma unroll
      for (int ni = 0; ni < 3; ++ni)
        acc[mi][ni] = WMMA16(a[mi], bf[ni], acc[mi][ni]);
  }
#endif
#pragma unroll
  for (int mi = 0; mi < 3; ++mi)
#pragma unroll
    for (int ni = 0; ni < 3; ++ni)
#pragma unroll
      for (int i = 0; i < 8; ++i)
        atomicAdd(&C[(mi * 16 + i + 8 * sub) * chead + ni * 16 + nl], acc[mi][ni][i]);
}

// ---------------------------------------------------------------------------
// Batched GEMM (NN): out[z] = att[z](48x48) x v[z](48 x n).
// ---------------------------------------------------------------------------
__global__ __launch_bounds__(32)
void k_gemm_nn_wmma(const float* __restrict__ att, const float* __restrict__ v,
                    float* __restrict__ out,
                    int heads, int chead, int n,
                    int CvFull, int cv0, int CoFull, int co0)
{
  const int z = blockIdx.z;
  const int b = z / heads, h = z - b * heads;
  const float* A  = att + (size_t)z * chead * chead;
  const float* Bv = v   + ((size_t)b * CvFull + cv0 + h * chead) * n;
  float* C        = out + ((size_t)b * CoFull + co0 + h * chead) * n;
  const int lane = (int)threadIdx.x, nl = lane & 15, sub = lane >> 4;
  const int n0 = blockIdx.x * 16;

  v8f acc[3] = {v8f{}, v8f{}, v8f{}};
  for (int k0 = 0; k0 < chead; k0 += 32) {
    v16h bf;
#pragma unroll
    for (int j = 0; j < 16; ++j) {
      int k = k0 + ((j & 8) << 1) + sub * 8 + (j & 7);
      bf[j] = (_Float16)((k < chead) ? Bv[(size_t)k * n + n0 + nl] : 0.f);
    }
#pragma unroll
    for (int mi = 0; mi < 3; ++mi) {
      const float* arow = A + (mi * 16 + nl) * chead;
      v16h a;
#pragma unroll
      for (int j = 0; j < 8; ++j) {
        int ka = k0 + sub * 8 + j, kb = k0 + 16 + sub * 8 + j;
        a[j]     = (_Float16)((ka < chead) ? arow[ka] : 0.f);
        a[8 + j] = (_Float16)((kb < chead) ? arow[kb] : 0.f);
      }
      acc[mi] = WMMA16(a, bf, acc[mi]);
    }
  }
#pragma unroll
  for (int mi = 0; mi < 3; ++mi)
#pragma unroll
    for (int i = 0; i < 8; ++i)
      C[(size_t)(mi * 16 + i + 8 * sub) * n + n0 + nl] = acc[mi][i];
}

// --------------------------- pointwise kernels -----------------------------
__global__ void k_ln(const float* __restrict__ x, const float* __restrict__ w,
                     const float* __restrict__ b, float* __restrict__ y,
                     int B, int CFull, int c0, int C, int HW)
{
  int p = blockIdx.x * blockDim.x + threadIdx.x;
  int NP = B * HW;
  if (p >= NP) return;
  int nb = p / HW, rem = p - nb * HW;
  const float* xp = x + ((size_t)nb * CFull + c0) * HW + rem;
  float mu = 0.f;
  for (int c = 0; c < C; ++c) mu += xp[(size_t)c * HW];
  mu /= (float)C;
  float var = 0.f;
  for (int c = 0; c < C; ++c) { float d = xp[(size_t)c * HW] - mu; var += d * d; }
  var /= (float)C;
  float rs = rsqrtf(var + 1e-5f);
  float* yp = y + (size_t)nb * C * HW + rem;
  for (int c = 0; c < C; ++c)
    yp[(size_t)c * HW] = (xp[(size_t)c * HW] - mu) * rs * w[c] + b[c];
}

__global__ void k_dw3(const float* __restrict__ x, const float* __restrict__ w,
                      float* __restrict__ y, int B, int C, int H, int W)
{
  long long i = (long long)blockIdx.x * blockDim.x + threadIdx.x;
  long long tot = (long long)B * C * H * W;
  if (i >= tot) return;
  int wv = (int)(i % W); long long t = i / W;
  int hv = (int)(t % H); t /= H;
  int c  = (int)(t % C);
  const float* xp = x + (i - hv * W - wv);
  const float* wp = w + c * 9;
  float acc = 0.f;
#pragma unroll
  for (int kh = 0; kh < 3; ++kh) {
    int ih = hv + kh - 1;
    if (ih < 0 || ih >= H) continue;
#pragma unroll
    for (int kw = 0; kw < 3; ++kw) {
      int iw = wv + kw - 1;
      if (iw < 0 || iw >= W) continue;
      acc += xp[ih * W + iw] * wp[kh * 3 + kw];
    }
  }
  y[i] = acc;
}

__global__ void k_rownorm(float* __restrict__ x, int CFull, int c0, int C, int n)
{
  int r = blockIdx.x;
  int b = r / C, c = r - b * C;
  float* row = x + ((size_t)b * CFull + c0 + c) * n;
  float s = 0.f;
  for (int i = threadIdx.x; i < n; i += blockDim.x) { float v = row[i]; s += v * v; }
  __shared__ float red[256];
  red[threadIdx.x] = s;
  __syncthreads();
  for (int st = 128; st > 0; st >>= 1) {
    if ((int)threadIdx.x < st) red[threadIdx.x] += red[threadIdx.x + st];
    __syncthreads();
  }
  float inv = 1.f / fmaxf(sqrtf(red[0]), 1e-12f);
  for (int i = threadIdx.x; i < n; i += blockDim.x) row[i] *= inv;
}

__global__ void k_softmax_rows(float* __restrict__ a, const float* __restrict__ temp,
                               int heads, int chead, int rows)
{
  int r = blockIdx.x * blockDim.x + threadIdx.x;
  if (r >= rows) return;
  float* row = a + (size_t)r * chead;
  float t = temp[(r / chead) % heads];
  float mx = -3.4e38f;
  for (int j = 0; j < chead; ++j) { row[j] *= t; mx = fmaxf(mx, row[j]); }
  float s = 0.f;
  for (int j = 0; j < chead; ++j) { float e = expf(row[j] - mx); row[j] = e; s += e; }
  float inv = 1.f / s;
  for (int j = 0; j < chead; ++j) row[j] *= inv;
}

__global__ void k_gelu_mul(const float* __restrict__ x, float* __restrict__ y,
                           int B, int hid, int HW)
{
  long long i = (long long)blockIdx.x * blockDim.x + threadIdx.x;
  long long tot = (long long)B * hid * HW;
  if (i >= tot) return;
  int p = (int)(i % HW); long long t = i / HW;
  int c = (int)(t % hid);
  int b = (int)(t / hid);
  float x1 = x[((size_t)b * 2 * hid + c) * HW + p];
  float x2 = x[((size_t)b * 2 * hid + hid + c) * HW + p];
  float g = 0.5f * x1 * (1.f + erff(x1 * 0.70710678118654752f));
  y[i] = g * x2;
}

__global__ void k_add(float* __restrict__ o, const float* __restrict__ a,
                      const float* __restrict__ b, long long n)
{
  long long i = (long long)blockIdx.x * blockDim.x + threadIdx.x;
  if (i < n) o[i] = a[i] + b[i];
}

__global__ void k_copy(float* __restrict__ o, const float* __restrict__ a, long long n)
{
  long long i = (long long)blockIdx.x * blockDim.x + threadIdx.x;
  if (i < n) o[i] = a[i];
}

__global__ void k_zero(float* __restrict__ o, long long n)
{
  long long i = (long long)blockIdx.x * blockDim.x + threadIdx.x;
  if (i < n) o[i] = 0.f;
}

__global__ void k_up2(const float* __restrict__ x, float* __restrict__ y,
                      int B, int C, int H, int W)
{
  long long i = (long long)blockIdx.x * blockDim.x + threadIdx.x;
  long long tot = (long long)B * C * 4 * H * W;
  if (i >= tot) return;
  int W2 = 2 * W;
  int wv = (int)(i % W2); long long t = i / W2;
  int hv = (int)(t % (2 * H)); t /= (2 * H);
  y[i] = x[t * H * W + (hv >> 1) * W + (wv >> 1)];
}

__global__ void k_chcopy(const float* __restrict__ s, float* __restrict__ d,
                         int B, int Cs, int HW, int Cd, int c0)
{
  long long i = (long long)blockIdx.x * blockDim.x + threadIdx.x;
  long long tot = (long long)B * Cs * HW;
  if (i >= tot) return;
  int p = (int)(i % HW); long long t = i / HW;
  int c = (int)(t % Cs);
  int b = (int)(t / Cs);
  d[((size_t)b * Cd + c0 + c) * HW + p] = s[i];
}

struct Blk {
  const float *n1w, *n1b, *n2w, *n2b, *n3w, *n3b, *temp;
  const float *q, *q_dw, *kv, *kv_dw, *apo, *pi, *dw, *fpo;
};
struct PW { const _Float16* p; int CinP; };

extern "C" void kernel_launch(void* const* d_in, const int* in_sizes, int n_in,
                              void* d_out, int out_size, void* d_ws, size_t ws_size,
                              hipStream_t stream)
{
  (void)in_sizes; (void)n_in; (void)out_size; (void)ws_size;
  const int B = 2;
  const float* X     = (const float*)d_in[0];
  const float* PYR0  = (const float*)d_in[1];
  const float* PYR1  = (const float*)d_in[2];
  const float* PYR2  = (const float*)d_in[3];
  const float* FAKE  = (const float*)d_in[4];
  const float* REALA = (const float*)d_in[5];
  int ii = 6;
  auto P = [&]() { return (const float*)d_in[ii++]; };
  const float* init_w = P(); const float* init_b = P();
  const float *rw1[3], *rb1[3], *rw2[3], *rb2[3];
  for (int r = 0; r < 3; ++r) { rw1[r] = P(); rb1[r] = P(); rw2[r] = P(); rb2[r] = P(); }
  const float* out_w = P(); const float* out_b = P();
  const float* pos_w = P(); const float* pos_b = P();
  const float* pe1[3]; pe1[0] = P(); pe1[1] = P(); pe1[2] = P();
  const float* pe2[2]; pe2[0] = P(); pe2[1] = P();
  const float* pe3[1]; pe3[0] = P();
  Blk bk[3];
  for (int t = 0; t < 3; ++t) {
    bk[t].n1w = P(); bk[t].n1b = P(); bk[t].n2w = P(); bk[t].n2b = P();
    bk[t].n3w = P(); bk[t].n3b = P(); bk[t].temp = P();
    bk[t].q = P(); bk[t].q_dw = P(); bk[t].kv = P(); bk[t].kv_dw = P();
    bk[t].apo = P(); bk[t].pi = P(); bk[t].dw = P(); bk[t].fpo = P();
  }
  const float* proj[3]; proj[0] = P(); proj[1] = P(); proj[2] = P();

  float* out = (float*)d_out;
  float* o_hf3 = out;                       // (2,3,256,256)
  float* o_hf2 = out + 393216;              // (2,3,128,128)
  float* o_hf1 = out + 393216 + 98304;      // (2,3,64,64)
  float* o_fl  = out + 516096;              // (2,3,32,32)

  // ---- workspace bump allocator --------------------------------------
  size_t wpos = 0;
  float* ws = (float*)d_ws;
  auto alloc = [&](size_t nf) { nf = (nf + 3) & ~(size_t)3; float* p = ws + wpos; wpos += nf; return p; };

  // ---- weight repack (persistent, done once per launch) ---------------
  auto pack = [&](const float* w, int Cout, int Cin, int KK) -> PW {
    int CoutP = (Cout + 31) & ~31;
    int CinP  = (Cin + 31) & ~31;
    long long tot = (long long)CoutP * KK * CinP;
    _Float16* dst = (_Float16*)alloc((size_t)(tot / 2 + 4));
    k_wpack<<<dim3((unsigned)((tot + 255) / 256)), 256, 0, stream>>>(w, dst, Cout, CoutP, Cin, CinP, KK);
    PW r; r.p = dst; r.CinP = CinP; return r;
  };
  PW pk_pos  = pack(pos_w, 64, 3, 9);
  PW pk_init = pack(init_w, 64, 9, 9);
  PW pk_rw1[3], pk_rw2[3];
  for (int r = 0; r < 3; ++r) { pk_rw1[r] = pack(rw1[r], 64, 64, 9); pk_rw2[r] = pack(rw2[r], 64, 64, 9); }
  PW pk_out = pack(out_w, 3, 64, 9);
  PW pk_pe1[3] = { pack(pe1[0], 96, 6, 9), pack(pe1[1], 192, 96, 9), pack(pe1[2], 384, 192, 9) };
  PW pk_pe2[2] = { pack(pe2[0], 96, 6, 9), pack(pe2[1], 192, 96, 9) };
  PW pk_pe3[1] = { pack(pe3[0], 96, 6, 9) };
  const int dms[3] = {192, 96, 48}, hids[3] = {510, 255, 127};
  PW pk_q[3], pk_kv[3], pk_apo[3], pk_pi[3], pk_fpo[3], pk_proj[3];
  for (int t = 0; t < 3; ++t) {
    int d = dms[t], hd = hids[t];
    pk_q[t]    = pack(bk[t].q,   d, d, 1);
    pk_kv[t]   = pack(bk[t].kv,  2 * d, d, 1);
    pk_apo[t]  = pack(bk[t].apo, d, d, 1);
    pk_pi[t]   = pack(bk[t].pi,  2 * hd, d, 1);
    pk_fpo[t]  = pack(bk[t].fpo, d, hd, 1);
    pk_proj[t] = pack(proj[t], 3, d, 9);
  }

  // ---- launch helpers -------------------------------------------------
  auto conv = [&](const float* xin, PW w, const float* bias,
                  const float* res, float* yout, int Cin, int H, int Wd,
                  int Cout, int KH, int pad, int act) {
    dim3 g((unsigned)((B * H * Wd + 63) / 64), (unsigned)((Cout + 31) / 32), 1);
    k_conv_wmma<<<g, 32, 0, stream>>>(xin, w.p, bias, res, yout,
                                      B, Cin, w.CinP, H, Wd, Cout, KH, KH, pad, act);
  };
  auto add = [&](float* o, const float* a, const float* b, long long n) {
    k_add<<<dim3((unsigned)((n + 255) / 256)), 256, 0, stream>>>(o, a, b, n);
  };

  // =====================================================================
  // mask branch (B,64,64,64)
  // =====================================================================
  const int H0 = 64, HW0 = H0 * H0;
  float* pos  = alloc((size_t)B * 64 * HW0);
  float* mA   = alloc((size_t)B * 64 * HW0);
  float* mB   = alloc((size_t)B * 64 * HW0);
  float* tRB  = alloc((size_t)B * 64 * HW0);
  float* mask64  = alloc((size_t)B * 3 * HW0);
  float* mask128 = alloc((size_t)B * 3 * 4 * HW0);
  float* mask256 = alloc((size_t)B * 3 * 16 * HW0);
  const size_t level_base = wpos;

  conv(REALA, pk_pos, pos_b, nullptr, pos, 3, H0, H0, 64, 3, 1, 0);
  conv(X, pk_init, init_b, pos, mA, 9, H0, H0, 64, 3, 1, 1);   // lrelu(conv)+pos
  float* cm = mA; float* nm = mB;
  for (int r = 0; r < 3; ++r) {
    conv(cm, pk_rw1[r], rb1[r], nullptr, tRB, 64, H0, H0, 64, 3, 1, 1);   // lrelu
    conv(tRB, pk_rw2[r], rb2[r], cm, nm, 64, H0, H0, 64, 3, 1, 0);        // + skip
    add(nm, nm, pos, (long long)B * 64 * HW0);                            // + pos
    float* sw = cm; cm = nm; nm = sw;
  }
  conv(cm, pk_out, out_b, nullptr, mask64, 64, H0, H0, 3, 3, 1, 0);

  // =====================================================================
  // one pyramid level
  // =====================================================================
  auto run_level = [&](const float* pyr, const float* maskL, int H,
                       const PW* pe_pk, const int* pe_c, int npe,
                       const Blk& bkp, PW pq, PW pkv, PW papo, PW ppi, PW pfpo,
                       PW pproj, int heads, int hid, float* hf_out) {
    wpos = level_base;
    const int HW = H * H;
    const long long EW = (long long)B * HW;
    // concat [pyr, mask]
    float* t0 = alloc((size_t)B * 6 * HW);
    k_chcopy<<<dim3((unsigned)((EW * 3 + 255) / 256)), 256, 0, stream>>>(pyr, t0, B, 3, HW, 6, 0);
    k_chcopy<<<dim3((unsigned)((EW * 3 + 255) / 256)), 256, 0, stream>>>(maskL, t0, B, 3, HW, 6, 3);
    // patch-embed chain
    const float* cur = t0; int cc = 6;
    for (int j = 0; j < npe; ++j) {
      float* nt = alloc((size_t)B * pe_c[j] * HW);
      conv(cur, pe_pk[j], nullptr, nullptr, nt, cc, H, H, pe_c[j], 3, 1, 0);
      cur = nt; cc = pe_c[j];
    }
    const int dim = cc / 2, ch = dim / heads;       // ch == 48 at every level
    float* n1 = alloc((size_t)B * dim * HW);
    float* n2 = alloc((size_t)B * dim * HW);
    k_ln<<<dim3((unsigned)((EW + 255) / 256)), 256, 0, stream>>>(cur, bkp.n1w, bkp.n1b, n1, B, cc, 0, dim, HW);
    k_ln<<<dim3((unsigned)((EW + 255) / 256)), 256, 0, stream>>>(cur, bkp.n2w, bkp.n2b, n2, B, cc, dim, dim, HW);
    float* q1  = alloc((size_t)B * dim * HW);
    float* Q   = alloc((size_t)B * dim * HW);
    float* kv1 = alloc((size_t)B * 2 * dim * HW);
    float* KV  = alloc((size_t)B * 2 * dim * HW);
    float* att = alloc((size_t)B * heads * ch * ch);
    conv(n1, pq, nullptr, nullptr, q1, dim, H, H, dim, 1, 0, 0);
    k_dw3<<<dim3((unsigned)((EW * dim + 255) / 256)), 256, 0, stream>>>(q1, bkp.q_dw, Q, B, dim, H, H);
    conv(n2, pkv, nullptr, nullptr, kv1, dim, H, H, 2 * dim, 1, 0, 0);
    k_dw3<<<dim3((unsigned)((EW * 2 * dim + 255) / 256)), 256, 0, stream>>>(kv1, bkp.kv_dw, KV, B, 2 * dim, H, H);
    k_rownorm<<<dim3((unsigned)(B * dim)), 256, 0, stream>>>(Q, dim, 0, dim, HW);
    k_rownorm<<<dim3((unsigned)(B * dim)), 256, 0, stream>>>(KV, 2 * dim, 0, dim, HW);
    // a = q k^T (48x48 per head, K = HW), split-K with atomic accumulation
    long long attN = (long long)B * heads * ch * ch;
    k_zero<<<dim3((unsigned)((attN + 255) / 256)), 256, 0, stream>>>(att, attN);
    int nsplit = (HW >= 65536) ? 16 : ((HW >= 16384) ? 8 : 4);   // chunk = mult of 128
    k_gemm_nt_wmma<<<dim3((unsigned)nsplit, 1, (unsigned)(B * heads)), 32, 0, stream>>>(
        Q, KV, att, heads, ch, HW, dim, 0, 2 * dim, 0, nsplit);
    int rows = B * heads * ch;
    k_softmax_rows<<<dim3((unsigned)((rows + 63) / 64)), 64, 0, stream>>>(att, bkp.temp, heads, ch, rows);
    float* O = q1;                                   // reuse
    k_gemm_nn_wmma<<<dim3((unsigned)(HW / 16), 1, (unsigned)(B * heads)), 32, 0, stream>>>(
        att, KV, O, heads, ch, HW, 2 * dim, dim, dim, 0);
    float* Y = kv1;                                  // reuse
    conv(O, papo, nullptr, n1, Y, dim, H, H, dim, 1, 0, 0);
    add(Y, Y, n1, EW * dim);
    float* N3 = kv1 + (size_t)B * dim * HW;          // second half of kv1 region
    k_ln<<<dim3((unsigned)((EW + 255) / 256)), 256, 0, stream>>>(Y, bkp.n3w, bkp.n3b, N3, B, dim, 0, dim, HW);
    float* P1 = alloc((size_t)B * 2 * hid * HW);
    float* P2 = alloc((size_t)B * 2 * hid * HW);
    conv(N3, ppi, nullptr, nullptr, P1, dim, H, H, 2 * hid, 1, 0, 0);
    k_dw3<<<dim3((unsigned)((EW * 2 * hid + 255) / 256)), 256, 0, stream>>>(P1, bkp.dw, P2, B, 2 * hid, H, H);
    float* G = P1;                                   // reuse
    k_gelu_mul<<<dim3((unsigned)((EW * hid + 255) / 256)), 256, 0, stream>>>(P2, G, B, hid, HW);
    float* Y2 = Q;                                   // reuse
    conv(G, pfpo, nullptr, Y, Y2, hid, H, H, dim, 1, 0, 0);
    conv(Y2, pproj, nullptr, maskL, hf_out, dim, H, H, 3, 3, 1, 0);
  };

  const int c1[3] = {96, 192, 384};
  const int c2[2] = {96, 192};
  const int c3[1] = {96};
  run_level(PYR2, mask64, 64, pk_pe1, c1, 3, bk[0], pk_q[0], pk_kv[0], pk_apo[0],
            pk_pi[0], pk_fpo[0], pk_proj[0], 4, 510, o_hf1);
  k_up2<<<dim3((unsigned)((B * 3 * 4 * HW0 + 255) / 256)), 256, 0, stream>>>(mask64, mask128, B, 3, 64, 64);
  run_level(PYR1, mask128, 128, pk_pe2, c2, 2, bk[1], pk_q[1], pk_kv[1], pk_apo[1],
            pk_pi[1], pk_fpo[1], pk_proj[1], 2, 255, o_hf2);
  k_up2<<<dim3((unsigned)((B * 3 * 16 * HW0 + 255) / 256)), 256, 0, stream>>>(mask128, mask256, B, 3, 128, 128);
  run_level(PYR0, mask256, 256, pk_pe3, c3, 1, bk[2], pk_q[2], pk_kv[2], pk_apo[2],
            pk_pi[2], pk_fpo[2], pk_proj[2], 1, 127, o_hf3);

  k_copy<<<dim3((unsigned)((6144 + 255) / 256)), 256, 0, stream>>>(o_fl, FAKE, 6144);
}
